// acLSTM_11862699671612
// MI455X (gfx1250) — matrix operand order repaired and checked by reading, not compiled, not run on hardware
//
#include <hip/hip_runtime.h>
#include <hip/hip_bf16.h>
#include <stdint.h>

// ---------------- types ----------------
typedef __bf16 bf16;
typedef __attribute__((ext_vector_type(16))) __bf16 v16bf;
typedef __attribute__((ext_vector_type(8)))  float  v8f;
typedef __attribute__((ext_vector_type(4)))  float  f32x4;

#define HSZ   1024
#define BATCH 256
#define TSTEPS 100
#define INDIM 132
#define OUTDIM 132
#define XPAD  160          // 132 padded up to a multiple of 32
#define HB    (BATCH * HSZ) // elements per (layer) state buffer

// ============================================================
// Weight packing: W (4H x Ksrc, f32, gate blocks i,f,g,o) ->
// bf16 tiles in exact WMMA-B lane order.
// Packed column p = 4*j + gate  (so each 256-wide N block owns full
// i/f/g/o quadruples for 64 hidden units).
// Tile (ntile, ktile): 512 bf16; lane l holds 16 k values of packed row
// ntile*16 + (l%16): k = ktile*32 + 8*(l/16) + {0..7, 16..23}.
// ============================================================
__global__ void pack_weight(const float* __restrict__ src, bf16* __restrict__ dst,
                            int Kpad, int Ksrc) {
  int idx = blockIdx.x * 256 + threadIdx.x;
  int total = 4096 * Kpad;
  if (idx >= total) return;
  int e    = idx & 15;
  int lane = (idx >> 4) & 31;
  int tIdx = idx >> 9;
  int K32  = Kpad >> 5;
  int ktile = tIdx % K32;
  int ntile = tIdx / K32;
  int p = ntile * 16 + (lane & 15);
  int k = ktile * 32 + ((lane >> 4) << 3) + (e < 8 ? e : e + 8);
  int j = p >> 2, g = p & 3;
  float v = (k < Ksrc) ? src[(size_t)(g * HSZ + j) * Ksrc + k] : 0.0f;
  dst[idx] = (bf16)v;
}

__global__ void pack_bias(const float* __restrict__ bih, const float* __restrict__ bhh,
                          float* __restrict__ dst) {
  int idx = blockIdx.x * 256 + threadIdx.x;
  if (idx >= 4096) return;
  int j = idx >> 2, g = idx & 3;
  dst[idx] = bih[g * HSZ + j] + bhh[g * HSZ + j];
}

// real_seq [B,T,132] f32 -> Xgt [T,B,160] bf16 (zero-padded)
__global__ void pack_xgt(const float* __restrict__ seq, bf16* __restrict__ dst) {
  int idx = blockIdx.x * 256 + threadIdx.x;
  if (idx >= TSTEPS * BATCH * XPAD) return;
  int k = idx % XPAD;
  int b = (idx / XPAD) % BATCH;
  int t = idx / (XPAD * BATCH);
  float v = (k < INDIM) ? seq[((size_t)b * TSTEPS + t) * INDIM + k] : 0.0f;
  dst[idx] = (bf16)v;
}

__global__ void zero_state(bf16* __restrict__ h0, float* __restrict__ c) {
  int idx = blockIdx.x * 256 + threadIdx.x;
  if (idx >= 3 * HB) return;
  h0[idx] = (bf16)0.0f;
  c[idx]  = 0.0f;
}

// ============================================================
// Fused 3-cell LSTM step kernel.
// grid = (16, 4, 3): x -> 256 packed gate cols (64 hidden units),
//                    y -> 64 batch rows, z -> layer.
// block = 256 threads = 8 wave32 waves, wave tile = 32(M) x 64(N).
// A operands are loaded straight from global memory in WMMA lane
// layout (two b128 per operand) -> no barriers in the K loop; waves
// sharing rows hit the same lines in WGP$/L2.
// ============================================================
struct CellArgs {
  const bf16*  X[3];      // layer input (bf16): xframe / h0_prev / h1_prev
  const bf16*  Wih[3];    // packed bf16 tiles
  const bf16*  Whh[3];
  const float* bias[3];   // packed (4*j+gate) combined bias
  const bf16*  Hprev[3];
  bf16*        Hnext[3];
  float*       C[3];
  int          Kx[3];     // padded K (and row stride) of X part
};

__global__ __launch_bounds__(256) void lstm_cell_kernel(CellArgs args) {
  const int l    = blockIdx.z;
  const int tid  = threadIdx.x;
  const int lane = tid & 31;
  const int wave = tid >> 5;
  const int waveM = wave & 1;   // 2 waves over M (32 rows each)
  const int waveN = wave >> 1;  // 4 waves over N (64 cols each)
  const int m0 = blockIdx.y * 64;

  extern __shared__ char smem[];
  float* sG = (float*)smem;     // [64][256] f32 gates

  const int half = lane >> 4;   // selects k sub-pattern per WMMA A layout
  const int mrow = lane & 15;
  const int rbase = m0 + waveM * 32 + mrow;   // this lane's A row (mi=0)

  v8f acc[2][4] = {};

  for (int phase = 0; phase < 2; ++phase) {
    const bf16* Asrc = (phase == 0) ? args.X[l]   : args.Hprev[l];
    const bf16* Wpk  = (phase == 0) ? args.Wih[l] : args.Whh[l];
    const int   K    = (phase == 0) ? args.Kx[l]  : HSZ;
    const int   K32  = K >> 5;
    const int   ntbase = blockIdx.x * 16 + waveN * 4;
    // per-lane A row pointers (k advances by 32 per tile)
    const bf16* arow0 = Asrc + (size_t)rbase * K + half * 8;
    const bf16* arow1 = arow0 + (size_t)16 * K;

    for (int kt = 0; kt < K32; ++kt) {
      // ---- A operands straight from global (WMMA 16-bit A 16x32 layout) ----
      v16bf a[2];
      {
        union { f32x4 f[2]; v16bf v; } u0, u1;
        const bf16* p0 = arow0 + kt * 32;
        const bf16* p1 = arow1 + kt * 32;
        u0.f[0] = *reinterpret_cast<const f32x4*>(p0);        // k 0..7 / 8..15
        u0.f[1] = *reinterpret_cast<const f32x4*>(p0 + 16);   // k 16..23 / 24..31
        u1.f[0] = *reinterpret_cast<const f32x4*>(p1);
        u1.f[1] = *reinterpret_cast<const f32x4*>(p1 + 16);
        a[0] = u0.v;
        a[1] = u1.v;
      }

      // ---- B tiles from pre-packed weights + WMMA ----
#pragma unroll
      for (int ni = 0; ni < 4; ++ni) {
        const bf16* tb = Wpk + ((size_t)(ntbase + ni) * K32 + kt) * 512 + lane * 16;
        __builtin_prefetch((const void*)(tb + 512), 0, 1);  // next k-tile
        union { f32x4 f[2]; v16bf v; } ub;
        ub.f[0] = *reinterpret_cast<const f32x4*>(tb);
        ub.f[1] = *reinterpret_cast<const f32x4*>(tb + 8);
        acc[0][ni] = __builtin_amdgcn_wmma_f32_16x16x32_bf16(
            false, a[0], false, ub.v, (short)0, acc[0][ni], false, false);
        acc[1][ni] = __builtin_amdgcn_wmma_f32_16x16x32_bf16(
            false, a[1], false, ub.v, (short)0, acc[1][ni], false, false);
      }
    }
  }

  // ---- epilogue: bias, spill gates to LDS for cross-lane i/f/g/o ----
#pragma unroll
  for (int ni = 0; ni < 4; ++ni) {
    int nloc = waveN * 64 + ni * 16 + mrow;
    float bv = args.bias[l][blockIdx.x * 256 + nloc];
#pragma unroll
    for (int mi = 0; mi < 2; ++mi) {
      int mb = waveM * 32 + mi * 16 + half * 8;
      v8f v = acc[mi][ni];
#pragma unroll
      for (int q = 0; q < 8; ++q)
        sG[(size_t)(mb + q) * 256 + nloc] = v[q] + bv;
    }
  }
  __syncthreads();

  // ---- elementwise LSTM update: 64 rows x 64 hidden units ----
  const int j0 = blockIdx.x * 64;
  float* Cbuf = args.C[l];
  bf16*  Hn   = args.Hnext[l];
#pragma unroll 4
  for (int it = 0; it < 16; ++it) {
    int lin = tid + it * 256;
    int row = lin >> 6;
    int jl  = lin & 63;
    f32x4 g = *reinterpret_cast<const f32x4*>(sG + (size_t)row * 256 + jl * 4);
    float ig = 1.0f / (1.0f + __expf(-g.x));
    float fg = 1.0f / (1.0f + __expf(-g.y));
    float gg = tanhf(g.z);
    float og = 1.0f / (1.0f + __expf(-g.w));
    size_t off = (size_t)(m0 + row) * HSZ + (j0 + jl);
    float cn = fg * Cbuf[off] + ig * gg;
    Cbuf[off] = cn;
    Hn[off] = (bf16)(og * tanhf(cn));
  }
}

// ============================================================
// Decoder: out = h2 @ W_dec^T + b_dec  (256 x 132, K=1024, tiny)
// Writes d_out[b, t*132 + o] and the bf16 feedback frame (padded).
// ============================================================
__global__ void decoder_kernel(const bf16* __restrict__ h2,
                               const float* __restrict__ Wdec,
                               const float* __restrict__ bdec,
                               float* __restrict__ out,
                               bf16* __restrict__ xfeed, int t) {
  int idx = blockIdx.x * 256 + threadIdx.x;     // B*XPAD threads
  int b = idx / XPAD;
  int o = idx % XPAD;
  if (b >= BATCH) return;
  if (o < OUTDIM) {
    const bf16*  hp = h2 + (size_t)b * HSZ;
    const float* wp = Wdec + (size_t)o * HSZ;
    float acc = 0.0f;
    for (int k = 0; k < HSZ; k += 8) {
#pragma unroll
      for (int u = 0; u < 8; ++u) acc += (float)hp[k + u] * wp[k + u];
    }
    float r = acc + bdec[o];
    out[(size_t)b * TSTEPS * OUTDIM + (size_t)t * OUTDIM + o] = r;
    xfeed[(size_t)b * XPAD + o] = (bf16)r;
  } else {
    xfeed[(size_t)b * XPAD + o] = (bf16)0.0f;   // keep pad region zero
  }
}

// ============================================================
extern "C" void kernel_launch(void* const* d_in, const int* in_sizes, int n_in,
                              void* d_out, int out_size, void* d_ws, size_t ws_size,
                              hipStream_t stream) {
  (void)in_sizes; (void)n_in; (void)out_size; (void)ws_size;

  const float* real_seq = (const float*)d_in[0];
  const float* W_ih1 = (const float*)d_in[1];
  const float* W_hh1 = (const float*)d_in[2];
  const float* b_ih1 = (const float*)d_in[3];
  const float* b_hh1 = (const float*)d_in[4];
  const float* W_ih2 = (const float*)d_in[5];
  const float* W_hh2 = (const float*)d_in[6];
  const float* b_ih2 = (const float*)d_in[7];
  const float* b_hh2 = (const float*)d_in[8];
  const float* W_ih3 = (const float*)d_in[9];
  const float* W_hh3 = (const float*)d_in[10];
  const float* b_ih3 = (const float*)d_in[11];
  const float* b_hh3 = (const float*)d_in[12];
  const float* W_dec = (const float*)d_in[13];
  const float* b_dec = (const float*)d_in[14];

  // ---- workspace carve-out (256B aligned) ----
  char* ws = (char*)d_ws;
  size_t off = 0;
  auto carve = [&](size_t bytes) -> void* {
    void* p = ws + off;
    off = (off + bytes + 255) & ~(size_t)255;
    return p;
  };
  bf16*  Wih1p = (bf16*)carve((size_t)4096 * XPAD * 2);
  bf16*  Whh1p = (bf16*)carve((size_t)4096 * HSZ * 2);
  bf16*  Wih2p = (bf16*)carve((size_t)4096 * HSZ * 2);
  bf16*  Whh2p = (bf16*)carve((size_t)4096 * HSZ * 2);
  bf16*  Wih3p = (bf16*)carve((size_t)4096 * HSZ * 2);
  bf16*  Whh3p = (bf16*)carve((size_t)4096 * HSZ * 2);
  float* biasp = (float*)carve((size_t)3 * 4096 * 4);
  bf16*  Xgt   = (bf16*)carve((size_t)TSTEPS * BATCH * XPAD * 2);
  bf16*  Xfeed = (bf16*)carve((size_t)BATCH * XPAD * 2);
  bf16*  hbufA = (bf16*)carve((size_t)3 * HB * 2);
  bf16*  hbufB = (bf16*)carve((size_t)3 * HB * 2);
  float* Cst   = (float*)carve((size_t)3 * HB * 4);
  bf16*  hbuf[2] = {hbufA, hbufB};

  // ---- one-time (per call) packing / init ----
  pack_weight<<<(4096 * XPAD + 255) / 256, 256, 0, stream>>>(W_ih1, Wih1p, XPAD, INDIM);
  pack_weight<<<(4096 * HSZ  + 255) / 256, 256, 0, stream>>>(W_hh1, Whh1p, HSZ, HSZ);
  pack_weight<<<(4096 * HSZ  + 255) / 256, 256, 0, stream>>>(W_ih2, Wih2p, HSZ, HSZ);
  pack_weight<<<(4096 * HSZ  + 255) / 256, 256, 0, stream>>>(W_hh2, Whh2p, HSZ, HSZ);
  pack_weight<<<(4096 * HSZ  + 255) / 256, 256, 0, stream>>>(W_ih3, Wih3p, HSZ, HSZ);
  pack_weight<<<(4096 * HSZ  + 255) / 256, 256, 0, stream>>>(W_hh3, Whh3p, HSZ, HSZ);
  pack_bias<<<16, 256, 0, stream>>>(b_ih1, b_hh1, biasp + 0 * 4096);
  pack_bias<<<16, 256, 0, stream>>>(b_ih2, b_hh2, biasp + 1 * 4096);
  pack_bias<<<16, 256, 0, stream>>>(b_ih3, b_hh3, biasp + 2 * 4096);
  pack_xgt<<<(TSTEPS * BATCH * XPAD + 255) / 256, 256, 0, stream>>>(real_seq, Xgt);
  zero_state<<<(3 * HB + 255) / 256, 256, 0, stream>>>(hbuf[0], Cst);

  // ---- sequential timesteps ----
  for (int t = 0; t < TSTEPS; ++t) {
    bool use_gt = (t % 10) < 5;   // [1]*5 + [0]*5, tiled (gt=5, cond=5)
    int cur = t & 1;

    CellArgs ca;
    ca.X[0] = use_gt ? (Xgt + (size_t)t * BATCH * XPAD) : Xfeed;
    ca.Kx[0] = XPAD;
    ca.X[1] = hbuf[cur] + 0 * HB;  ca.Kx[1] = HSZ;   // old h0
    ca.X[2] = hbuf[cur] + 1 * HB;  ca.Kx[2] = HSZ;   // old h1
    ca.Wih[0] = Wih1p; ca.Whh[0] = Whh1p;
    ca.Wih[1] = Wih2p; ca.Whh[1] = Whh2p;
    ca.Wih[2] = Wih3p; ca.Whh[2] = Whh3p;
    for (int l = 0; l < 3; ++l) {
      ca.bias[l]  = biasp + (size_t)l * 4096;
      ca.Hprev[l] = hbuf[cur] + (size_t)l * HB;
      ca.Hnext[l] = hbuf[1 - cur] + (size_t)l * HB;
      ca.C[l]     = Cst + (size_t)l * HB;
    }

    lstm_cell_kernel<<<dim3(16, 4, 3), 256, 65536, stream>>>(ca);
    decoder_kernel<<<(BATCH * XPAD) / 256, 256, 0, stream>>>(
        hbuf[1 - cur] + 2 * HB, W_dec, b_dec, (float*)d_out, Xfeed, t);
  }
}
